// LuongAttention_49331994362117
// MI455X (gfx1250) — matrix operand order repaired and checked by reading, not compiled, not run on hardware
//
#include <hip/hip_runtime.h>
#include <hip/hip_bf16.h>

typedef __attribute__((ext_vector_type(16))) _Float16     v16h;
typedef __attribute__((ext_vector_type(8)))  float        v8f;
typedef __attribute__((ext_vector_type(4)))  float        v4f;
typedef __attribute__((ext_vector_type(4)))  unsigned int v4u;
typedef __attribute__((ext_vector_type(8)))  int          v8i;
typedef __attribute__((ext_vector_type(4)))  int          v4i;

#define BATCH 8
#define TQ    2048
#define TK    4096
#define DD    128
#define QTILE 16
#define NWAVE 8
#define VROW  20   // per-wave V staging row stride in floats (16 data + 4 pad)

// TDM builtins exist only in the device pass; host pass just needs to parse.
#if defined(__HIP_DEVICE_COMPILE__)
#if !(__has_builtin(__builtin_amdgcn_tensor_load_to_lds) && __has_builtin(__builtin_amdgcn_s_wait_tensorcnt))
#error "gfx1250 TDM builtins not available in device compile"
#endif
#define TDM_WAIT(n) __builtin_amdgcn_s_wait_tensorcnt(n)
#else
#define TDM_WAIT(n) ((void)0)
#endif

// Issue a TDM load of a 32x16 f32 column-slice of V from global into this
// wave's private LDS staging buffer. 2D descriptor: tile = [32 rows x 16 cols],
// row stride in memory = 128 floats, LDS pad 4 dwords per 16 dwords stored
// (row stride 20 floats -> bank spread).
__device__ __forceinline__ void tdm_issue_v_slice(const float* gsrc, unsigned lds_byte_off) {
  unsigned long long ga = (unsigned long long)(uintptr_t)gsrc;
  v4u g0;
  g0[0] = 1u;                                   // count=1 valid descriptor
  g0[1] = lds_byte_off;                         // D#.lds_addr
  g0[2] = (unsigned)ga;                         // D#.global_addr[31:0]
  g0[3] = (unsigned)((ga >> 32) & 0x01FFFFFFu)  // D#.global_addr[56:32]
        | (2u << 30);                           // type = 2 ("image")
  v8i g1;
  g1[0] = (int)((2u << 16)      // data_size = 4 bytes
              | (1u << 20)      // pad_enable
              | (3u << 22)      // pad_interval: 2^(3+1) = 16 dwords
              | (3u << 25));    // pad_amount: 3+1 = 4 dwords
  g1[1] = (int)(16u << 16);     // tensor_dim0[15:0] = 16 (slice width)
  g1[2] = (int)(4096u << 16);   // tensor_dim1[15:0] = 4096 (rows)
  g1[3] = (int)(16u << 16);     // tile_dim0 = 16
  g1[4] = 32;                   // tile_dim1 = 32
  g1[5] = 128;                  // tensor_dim0_stride = 128 elements (full V row)
  g1[6] = 0;
  g1[7] = 0;
  v4i z4 = {0, 0, 0, 0};
#if defined(__HIP_DEVICE_COMPILE__)
#if __clang_major__ >= 23
  v8i z8 = {0, 0, 0, 0, 0, 0, 0, 0};
  __builtin_amdgcn_tensor_load_to_lds(g0, g1, z4, z4, z8, 0);
#else
  __builtin_amdgcn_tensor_load_to_lds(g0, g1, z4, z4, 0);
#endif
#else
  (void)g0; (void)g1; (void)z4;   // host pass: stub
#endif
}

// One workgroup = one (batch, 16-query-row) tile.
// LDS: 16x4096 f32 score/prob strip (256 KB) + stats + 8 waves x 2 x (32x20) f32
// private V staging (~40 KB)  =>  ~296 KB of the 320 KB WGP budget.
__global__ __launch_bounds__(256) void luong_attn_kernel(
    const float* __restrict__ dec,   // [B, TQ, DD]
    const float* __restrict__ enc,   // [B, TK, DD]
    float* __restrict__ ctx,         // [B, TQ, DD]
    float* __restrict__ attn)        // [B, TQ, TK]
{
  extern __shared__ float smem[];
  float* S  = smem;                  // [QTILE][TK]
  float* mx = smem + QTILE * TK;     // [QTILE]
  float* rs = mx + QTILE;            // [QTILE]
  float* Vs = rs + QTILE;            // [NWAVE][2][32 * VROW] staging

  const int blk   = blockIdx.x;
  const int b     = blk / (TQ / QTILE);
  const int qt    = blk % (TQ / QTILE);
  const int qbase = qt * QTILE;

  const int tid  = threadIdx.x;
  const int wave = tid >> 5;
  const int lane = tid & 31;
  const int nrow = lane & 15;   // A-matrix M row / C-matrix N column
  const int grp  = lane >> 4;   // lane group (ISA 16-bit A layout)

  // ---- preload Q tile as f16 A-fragments (kept in VGPRs all kernel) ----
  v16h aq[4];
  {
    const float* qrow = dec + (size_t)(b * TQ + qbase + nrow) * DD;
#pragma unroll
    for (int ch = 0; ch < 4; ++ch) {
#pragma unroll
      for (int e = 0; e < 16; ++e) {
        // 16-bit A layout: half-slot e of lane (nrow,grp) holds K = (e&7)+8g+16(e>>3)
        int kl = (e & 7) + 8 * grp + 16 * (e >> 3);
        aq[ch][e] = (_Float16)qrow[ch * 32 + kl];
      }
    }
  }

  // ---- pass 1: S = Q * K^T, tiled over Tk, strip -> LDS ----
  for (int kt = wave; kt < TK / 16; kt += NWAVE) {
    const float* krows = enc + (size_t)(b * TK + kt * 16) * DD;
    __builtin_prefetch(enc + (size_t)(b * TK + (((kt + NWAVE) & (TK / 16 - 1)) * 16)) * DD, 0, 1);
    v8f c = {};
#pragma unroll
    for (int ch = 0; ch < 4; ++ch) {
      // B fragment: lane&15 = N (k column), half-slot e holds K = (e&7)+8g+16(e>>3)
      v16h bfr;
      const int d = ch * 32;
#pragma unroll
      for (int e = 0; e < 16; ++e)
        bfr[e] = (_Float16)krows[(size_t)nrow * DD + d + (e & 7) + 8 * grp + 16 * (e >> 3)];
      c = __builtin_amdgcn_wmma_f32_16x16x32_f16(false, aq[ch], false, bfr,
                                                 (short)0, c, false, false);
    }
    // C layout: VGPR r -> M = r + 8*grp, column N = lane&15
#pragma unroll
    for (int r = 0; r < 8; ++r) {
      int m = r + 8 * grp;
      S[m * TK + kt * 16 + nrow] = c[r];
    }
  }
  __syncthreads();

  // ---- softmax stats: each wave reduces 2 rows (wave32 shuffles, float4 LDS reads) ----
  for (int m = wave * 2; m < wave * 2 + 2; ++m) {
    const v4f* srow4 = (const v4f*)(S + m * TK);
    float lmax = -3.402823466e38f;
    for (int j = lane; j < TK / 4; j += 32) {
      v4f v = srow4[j];
      lmax = fmaxf(lmax, fmaxf(fmaxf(v[0], v[1]), fmaxf(v[2], v[3])));
    }
#pragma unroll
    for (int off = 16; off > 0; off >>= 1)
      lmax = fmaxf(lmax, __shfl_xor(lmax, off, 32));
    float lsum = 0.f;
    for (int j = lane; j < TK / 4; j += 32) {
      v4f v = srow4[j];
      lsum += __expf(v[0] - lmax) + __expf(v[1] - lmax)
            + __expf(v[2] - lmax) + __expf(v[3] - lmax);
    }
#pragma unroll
    for (int off = 16; off > 0; off >>= 1)
      lsum += __shfl_xor(lsum, off, 32);
    if (lane == 0) { mx[m] = lmax; rs[m] = 1.f / lsum; }
  }
  __syncthreads();

  // ---- normalize in LDS + stream attention output as b128 stores ----
  {
    v4f* S4 = (v4f*)S;
    v4f* A4 = (v4f*)(attn + (size_t)(b * TQ + qbase) * TK);  // strip layout == output layout
    for (int idx4 = tid; idx4 < QTILE * TK / 4; idx4 += 256) {
      int m = idx4 >> 10;               // (idx4*4) / 4096
      v4f v = S4[idx4], p;
#pragma unroll
      for (int t = 0; t < 4; ++t) p[t] = __expf(v[t] - mx[m]) * rs[m];
      S4[idx4] = p;
      A4[(size_t)idx4] = p;
    }
  }
  __syncthreads();

  // ---- pass 2: context = P * V; each wave owns one 16-wide d-tile.
  //      Per-wave TDM double-buffered staging of its 32x16 V slice:
  //      no inter-wave synchronization anywhere in this loop. ----
  {
    const int dt = wave;                               // 8 waves * 16 = 128 = DD
    const float* gsl = enc + (size_t)b * TK * DD + dt * 16;  // column-slice base
    float* vbuf = Vs + wave * (2 * 32 * VROW);         // this wave's two buffers
    v8f c2 = {};

    tdm_issue_v_slice(gsl, (unsigned)(uintptr_t)(void*)vbuf);
    for (int kt = 0; kt < TK / 32; ++kt) {
      const int p = kt & 1;
      if (kt + 1 < TK / 32) {
        tdm_issue_v_slice(gsl + (size_t)(kt + 1) * 32 * DD,
                          (unsigned)(uintptr_t)(void*)(vbuf + (1 - p) * 32 * VROW));
        TDM_WAIT(1);                   // current slice landed
      } else {
        TDM_WAIT(0);
      }
      v16h pa, bv;
#pragma unroll
      for (int e = 0; e < 16; ++e) {
        int kl = (e & 7) + 8 * grp + 16 * (e >> 3);
        pa[e] = (_Float16)S[nrow * TK + kt * 32 + kl];
      }
      const float* vrow = vbuf + p * 32 * VROW;
#pragma unroll
      for (int e = 0; e < 16; ++e)
        bv[e] = (_Float16)vrow[((e & 7) + 8 * grp + 16 * (e >> 3)) * VROW + nrow];
      c2 = __builtin_amdgcn_wmma_f32_16x16x32_f16(false, pa, false, bv,
                                                  (short)0, c2, false, false);
    }
    float* crow = ctx + (size_t)(b * TQ + qbase) * DD + dt * 16 + nrow;
#pragma unroll
    for (int r = 0; r < 8; ++r) {
      int m = r + 8 * grp;
      crow[(size_t)m * DD] = c2[r];
    }
  }
}

extern "C" void kernel_launch(void* const* d_in, const int* in_sizes, int n_in,
                              void* d_out, int out_size, void* d_ws, size_t ws_size,
                              hipStream_t stream) {
  (void)in_sizes; (void)n_in; (void)out_size; (void)d_ws; (void)ws_size;
  const float* dec = (const float*)d_in[0];   // [8, 2048, 128]
  const float* enc = (const float*)d_in[1];   // [8, 4096, 128]
  float* ctx  = (float*)d_out;                            // [8, 2048, 128]
  float* attn = ctx + (size_t)BATCH * TQ * DD;            // [8, 2048, 4096]

  const int    nblocks  = BATCH * (TQ / QTILE);           // 1024
  const size_t lds_size =
      ((size_t)QTILE * TK + 2 * QTILE + (size_t)NWAVE * 2 * 32 * VROW) * sizeof(float); // ~296 KB

  luong_attn_kernel<<<dim3(nblocks), dim3(256), lds_size, stream>>>(dec, enc, ctx, attn);
}
